// VisionExpertAttention_49855980372284
// MI455X (gfx1250) — matrix operand order repaired and checked by reading, not compiled, not run on hardware
//
#include <hip/hip_runtime.h>
#include <hip/hip_bf16.h>
#include <stdint.h>

#define HIDDEN   4096
#define NHEADS   32
#define HEAD_DIM 128
#define BATCH    2
#define SEQ      2048
#define TOKENS   (BATCH * SEQ)     // 4096
#define O3       (3 * HIDDEN)      // 12288
#define SCALE_Q  0.08838834764831845f   // 1/sqrt(128), folded into Q at RoPE time

typedef __attribute__((ext_vector_type(16))) __bf16       bfrag;
typedef __attribute__((ext_vector_type(8)))  float        v8f;
typedef __attribute__((ext_vector_type(4)))  unsigned int u32x4;

union Frag { bfrag f; u32x4 u[2]; };

__device__ __forceinline__ unsigned short f32_to_bf16(float x) {
  union { float f; unsigned int u; } v; v.f = x;
  unsigned int r = v.u + 0x7FFFu + ((v.u >> 16) & 1u);  // round-to-nearest-even
  return (unsigned short)(r >> 16);
}

__device__ __forceinline__ v8f wmma_bf16(const bfrag& a, const bfrag& b, v8f c) {
  return __builtin_amdgcn_wmma_f32_16x16x32_bf16(false, a, false, b, (short)0, c,
                                                 false, false);
}

// ---------------------------------------------------------------- fp32 -> bf16
__global__ void k_f32_to_bf16(const float* __restrict__ src,
                              unsigned short* __restrict__ dst, int n4) {
  int i = blockIdx.x * blockDim.x + threadIdx.x;
  if (i >= n4) return;
  float4 v = ((const float4*)src)[i];
  unsigned int lo = (unsigned int)f32_to_bf16(v.x) |
                    ((unsigned int)f32_to_bf16(v.y) << 16);
  unsigned int hi = (unsigned int)f32_to_bf16(v.z) |
                    ((unsigned int)f32_to_bf16(v.w) << 16);
  ((uint2*)dst)[i] = make_uint2(lo, hi);
}

// ---------------------------------------------------------------- vision mask
__global__ void k_vmf(const int* __restrict__ tt, float* __restrict__ vmf) {
  int t = blockIdx.x * blockDim.x + threadIdx.x;
  if (t >= TOKENS) return;
  int i = t & (SEQ - 1);
  bool vm = (i < SEQ - 1) && (tt[t] == 1) && (tt[t + 1] == 1);
  vmf[t] = vm ? 1.0f : 0.0f;
}

// ------------------------------------------------- routed dual-expert GEMM
// Y[t][o] = vmf[t]*(X·Wv^T) + (1-vmf[t])*(X·Wl^T)
// X: [T][K] bf16 row-major, W*: [O][K] bf16 row-major, Y: [T][O] fp32.
// One wave computes a 64x32 output tile for BOTH experts.
// Grid: blockIdx.x = m-tile (FAST axis -> consecutive blocks reuse the same
// 512 KB (2 experts x 32 rows) weight tile through L2), blockIdx.y = n-tile.
template <int K, int O>
__global__ __launch_bounds__(32, 1)
void k_gemm_moe(const unsigned short* __restrict__ X,
                const unsigned short* __restrict__ Wv,
                const unsigned short* __restrict__ Wl,
                const float* __restrict__ vmf,
                float* __restrict__ Y) {
  const int lane  = threadIdx.x;
  const int row16 = lane & 15;
  const int hf    = lane >> 4;
  const int mBase = blockIdx.x * 64;   // fast axis: weight-tile reuse in L2
  const int nBase = blockIdx.y * 32;

  const unsigned short* ap = X  + (size_t)(mBase + row16) * K + hf * 16;
  const unsigned short* wv = Wv + (size_t)(nBase + row16) * K + hf * 16;
  const unsigned short* wl = Wl + (size_t)(nBase + row16) * K + hf * 16;
  constexpr int MS = 16 * K;   // A row-tile stride (elements), compile-time
  constexpr int NS = 16 * K;   // B col-tile stride (elements), compile-time

  v8f acc[4][2][2] = {};       // [mi][expert][n]

  for (int k0 = 0; k0 < K; k0 += 32, ap += 32, wv += 32, wl += 32) {
    Frag A[4];
#pragma unroll
    for (int mi = 0; mi < 4; ++mi) {
      A[mi].u[0] = *(const u32x4*)(ap + mi * MS);
      A[mi].u[1] = *(const u32x4*)(ap + mi * MS + 8);
    }
    {
      Frag B[2];
#pragma unroll
      for (int n = 0; n < 2; ++n) {
        B[n].u[0] = *(const u32x4*)(wv + n * NS);
        B[n].u[1] = *(const u32x4*)(wv + n * NS + 8);
      }
#pragma unroll
      for (int mi = 0; mi < 4; ++mi)
#pragma unroll
        for (int n = 0; n < 2; ++n)
          acc[mi][0][n] = wmma_bf16(A[mi].f, B[n].f, acc[mi][0][n]);
    }
    {
      Frag B[2];
#pragma unroll
      for (int n = 0; n < 2; ++n) {
        B[n].u[0] = *(const u32x4*)(wl + n * NS);
        B[n].u[1] = *(const u32x4*)(wl + n * NS + 8);
      }
#pragma unroll
      for (int mi = 0; mi < 4; ++mi)
#pragma unroll
        for (int n = 0; n < 2; ++n)
          acc[mi][1][n] = wmma_bf16(A[mi].f, B[n].f, acc[mi][1][n]);
    }
  }

#pragma unroll
  for (int mi = 0; mi < 4; ++mi)
#pragma unroll
    for (int r = 0; r < 8; ++r) {
      int row = mBase + mi * 16 + r + 8 * hf;   // C-layout: M = r + 8*(lane/16)
      float g = vmf[row];
#pragma unroll
      for (int n = 0; n < 2; ++n) {
        int o = nBase + n * 16 + row16;         // C-layout: N = lane%16
        Y[(size_t)row * O + o] =
            g * acc[mi][0][n][r] + (1.0f - g) * acc[mi][1][n][r];
      }
    }
}

// ---------------------------------------------------------------- RoPE + split
// qkv fp32 [T][12288] -> Q,K bf16 [B,NH,SEQ,DIM] (Q pre-scaled by 1/sqrt(d)),
// V bf16 transposed [B,NH,DIM,SEQ].
__global__ void k_rope(const float* __restrict__ qkv,
                       const int* __restrict__ pos_ids,
                       unsigned short* __restrict__ Qo,
                       unsigned short* __restrict__ Ko,
                       unsigned short* __restrict__ Vt) {
  unsigned int tid = blockIdx.x * blockDim.x + threadIdx.x;  // B*SEQ*NH*64
  int d = tid & 63;
  int h = (tid >> 6) & 31;
  int l = (tid >> 11) & (SEQ - 1);
  int b = tid >> 22;

  size_t base = (size_t)(b * SEQ + l) * O3 + h * HEAD_DIM;
  float q1 = qkv[base + d],              q2 = qkv[base + d + 64];
  float k1 = qkv[base + HIDDEN + d],     k2 = qkv[base + HIDDEN + d + 64];
  float v1 = qkv[base + 2 * HIDDEN + d], v2 = qkv[base + 2 * HIDDEN + d + 64];

  float pos = (float)pos_ids[b * SEQ + l];
  float inv = __powf(10000.0f, -(float)d * (1.0f / 64.0f));
  float s, c;
  __sincosf(pos * inv, &s, &c);

  size_t qk = ((size_t)(b * NHEADS + h) * SEQ + l) * HEAD_DIM;
  Qo[qk + d]      = f32_to_bf16((q1 * c - q2 * s) * SCALE_Q);
  Qo[qk + d + 64] = f32_to_bf16((q2 * c + q1 * s) * SCALE_Q);
  Ko[qk + d]      = f32_to_bf16(k1 * c - k2 * s);
  Ko[qk + d + 64] = f32_to_bf16(k2 * c + k1 * s);

  size_t vt = ((size_t)(b * NHEADS + h) * HEAD_DIM + d) * SEQ + l;
  Vt[vt]            = f32_to_bf16(v1);
  Vt[vt + 64 * SEQ] = f32_to_bf16(v2);
}

// ---------------------------------------------------------------- attention
// One wave per (b, h, 16-query tile); flash-style streaming over 32-key blocks.
__global__ __launch_bounds__(32, 1)
void k_attn(const unsigned short* __restrict__ Q,
            const unsigned short* __restrict__ Kk,
            const unsigned short* __restrict__ Vt,
            unsigned short* __restrict__ Ctx) {  // [B,SEQ,HIDDEN] bf16
  __shared__ __align__(16) unsigned short Plds[16 * 32];

  const int lane  = threadIdx.x;
  const int row16 = lane & 15;
  const int hf    = lane >> 4;
  const int q0 = blockIdx.x * 16;
  const int h  = blockIdx.y;
  const int b  = blockIdx.z;
  const size_t bh = (size_t)(b * NHEADS + h);

  // Q fragments for all four 32-wide d-chunks (A layout)
  const unsigned short* qrow = Q + (bh * SEQ + q0 + row16) * HEAD_DIM;
  Frag qa[4];
#pragma unroll
  for (int dc = 0; dc < 4; ++dc) {
    qa[dc].u[0] = *(const u32x4*)(qrow + dc * 32 + hf * 16);
    qa[dc].u[1] = *(const u32x4*)(qrow + dc * 32 + hf * 16 + 8);
  }

  // K base: advance 32 rows (= 32*128 elements) per key-step
  const unsigned short* kr = Kk + (bh * SEQ + row16) * HEAD_DIM + hf * 16;
  // V base: advance 32 elements per key-step; n-tile stride 16*SEQ elements
  const unsigned short* vr = Vt + (bh * HEAD_DIM + row16) * SEQ + hf * 16;

  v8f ctx[8] = {};
  float rmax[8], rsum[8];
#pragma unroll
  for (int r = 0; r < 8; ++r) { rmax[r] = -3.0e38f; rsum[r] = 0.0f; }

  for (int key0 = 0; key0 < SEQ;
       key0 += 32, kr += 32 * HEAD_DIM, vr += 32) {
    // ---- S = Q · K^T for 2 key tiles of 16 (scale already folded into Q)
    v8f s0 = {}, s1 = {};
    {
      Frag bk[8];
#pragma unroll
      for (int dc = 0; dc < 4; ++dc) {
        bk[dc].u[0]     = *(const u32x4*)(kr + dc * 32);
        bk[dc].u[1]     = *(const u32x4*)(kr + dc * 32 + 8);
        bk[4 + dc].u[0] = *(const u32x4*)(kr + 16 * HEAD_DIM + dc * 32);
        bk[4 + dc].u[1] = *(const u32x4*)(kr + 16 * HEAD_DIM + dc * 32 + 8);
      }
#pragma unroll
      for (int dc = 0; dc < 4; ++dc) {
        s0 = wmma_bf16(qa[dc].f, bk[dc].f, s0);
        s1 = wmma_bf16(qa[dc].f, bk[4 + dc].f, s1);
      }
    }

    // ---- online softmax (row = r + 8*hf spans 16 lanes of this half-group)
    float corr[8];
#pragma unroll
    for (int r = 0; r < 8; ++r) {
      float m = fmaxf(s0[r], s1[r]);
#pragma unroll
      for (int off = 1; off < 16; off <<= 1)
        m = fmaxf(m, __shfl_xor(m, off, 32));
      float nm = fmaxf(rmax[r], m);
      corr[r]  = __expf(rmax[r] - nm);
      float p0 = __expf(s0[r] - nm);
      float p1 = __expf(s1[r] - nm);
      float ls = p0 + p1;
#pragma unroll
      for (int off = 1; off < 16; off <<= 1)
        ls += __shfl_xor(ls, off, 32);
      rsum[r] = rsum[r] * corr[r] + ls;
      rmax[r] = nm;
      int row = r + 8 * hf;                       // C layout row
      Plds[row * 32 + row16]      = f32_to_bf16(p0);
      Plds[row * 32 + 16 + row16] = f32_to_bf16(p1);
    }
#pragma unroll
    for (int n = 0; n < 8; ++n)
#pragma unroll
      for (int r = 0; r < 8; ++r) ctx[n][r] *= corr[r];

    // ---- relayout P: C layout (LDS) -> A layout (regs)
    __syncthreads();
    Frag pa;
    pa.u[0] = *(const u32x4*)(&Plds[row16 * 32 + hf * 16]);
    pa.u[1] = *(const u32x4*)(&Plds[row16 * 32 + hf * 16 + 8]);
    __syncthreads();

    // ---- ctx += P · V, V fragments batched in groups of 4
#pragma unroll
    for (int g = 0; g < 2; ++g) {
      Frag bv[4];
#pragma unroll
      for (int n = 0; n < 4; ++n) {
        bv[n].u[0] = *(const u32x4*)(vr + (g * 4 + n) * 16 * SEQ);
        bv[n].u[1] = *(const u32x4*)(vr + (g * 4 + n) * 16 * SEQ + 8);
      }
#pragma unroll
      for (int n = 0; n < 4; ++n)
        ctx[g * 4 + n] = wmma_bf16(pa.f, bv[n].f, ctx[g * 4 + n]);
    }
  }

  // ---- normalize and write ctx as bf16 [B, SEQ, HIDDEN]
#pragma unroll
  for (int r = 0; r < 8; ++r) {
    float inv = 1.0f / rsum[r];
    int row = q0 + r + 8 * hf;
    size_t obase = (size_t)(b * SEQ + row) * HIDDEN + h * HEAD_DIM;
#pragma unroll
    for (int n = 0; n < 8; ++n)
      Ctx[obase + n * 16 + row16] = f32_to_bf16(ctx[n][r] * inv);
  }
}

// ---------------------------------------------------------------- launch
extern "C" void kernel_launch(void* const* d_in, const int* in_sizes, int n_in,
                              void* d_out, int out_size, void* d_ws, size_t ws_size,
                              hipStream_t stream) {
  const float* hidden = (const float*)d_in[0];
  const int*   tt     = (const int*)d_in[1];
  const int*   pos    = (const int*)d_in[2];
  // d_in[3] = attention_mask: all zeros, never applied by the reference
  const float* wvq = (const float*)d_in[4];
  const float* wlq = (const float*)d_in[5];
  const float* wvd = (const float*)d_in[6];
  const float* wld = (const float*)d_in[7];
  float* out = (float*)d_out;

  char* ws = (char*)d_ws;
  size_t off = 0;
  auto walloc = [&](size_t bytes) -> void* {
    void* p = ws + off;
    off = (off + bytes + 255) & ~(size_t)255;
    return p;
  };

  unsigned short* hb   = (unsigned short*)walloc((size_t)TOKENS * HIDDEN * 2);
  unsigned short* wvqb = (unsigned short*)walloc((size_t)O3 * HIDDEN * 2);
  unsigned short* wlqb = (unsigned short*)walloc((size_t)O3 * HIDDEN * 2);
  unsigned short* wvdb = (unsigned short*)walloc((size_t)HIDDEN * HIDDEN * 2);
  unsigned short* wldb = (unsigned short*)walloc((size_t)HIDDEN * HIDDEN * 2);
  float*          qkvf = (float*)walloc((size_t)TOKENS * O3 * 4);
  unsigned short* qb   = (unsigned short*)walloc((size_t)TOKENS * HIDDEN * 2);
  unsigned short* kb   = (unsigned short*)walloc((size_t)TOKENS * HIDDEN * 2);
  unsigned short* vtb  = (unsigned short*)walloc((size_t)TOKENS * HIDDEN * 2);
  unsigned short* ctxb = (unsigned short*)walloc((size_t)TOKENS * HIDDEN * 2);
  float*          vmfp = (float*)walloc((size_t)TOKENS * 4);

  auto conv = [&](const float* s, unsigned short* d, size_t n) {
    int n4 = (int)(n / 4);
    k_f32_to_bf16<<<(n4 + 255) / 256, 256, 0, stream>>>(s, d, n4);
  };
  conv(hidden, hb,   (size_t)TOKENS * HIDDEN);
  conv(wvq,    wvqb, (size_t)O3 * HIDDEN);
  conv(wlq,    wlqb, (size_t)O3 * HIDDEN);
  conv(wvd,    wvdb, (size_t)HIDDEN * HIDDEN);
  conv(wld,    wldb, (size_t)HIDDEN * HIDDEN);

  k_vmf<<<TOKENS / 256, 256, 0, stream>>>(tt, vmfp);

  // QKV: [4096 x 4096] x [12288 x 4096]^T, dual expert (64x32 tiles per wave)
  // m-tiles on the fast grid axis for weight-tile reuse in L2.
  k_gemm_moe<HIDDEN, O3><<<dim3(TOKENS / 64, O3 / 32), 32, 0, stream>>>(
      hb, wvqb, wlqb, vmfp, qkvf);

  k_rope<<<(BATCH * SEQ * NHEADS * 64) / 256, 256, 0, stream>>>(
      qkvf, pos, qb, kb, vtb);

  k_attn<<<dim3(SEQ / 16, NHEADS, BATCH), 32, 0, stream>>>(qb, kb, vtb, ctxb);

  // dense out: [4096 x 4096] x [4096 x 4096]^T, dual expert, writes d_out fp32
  k_gemm_moe<HIDDEN, HIDDEN><<<dim3(TOKENS / 64, HIDDEN / 32), 32, 0, stream>>>(
      ctxb, wvdb, wldb, vmfp, out);
}
